// cox_loss_v2_48249662603626
// MI455X (gfx1250) — matrix operand order repaired and checked by reading, not compiled, not run on hardware
//
#include <hip/hip_runtime.h>

typedef __attribute__((ext_vector_type(16))) _Float16 v16h;
typedef __attribute__((ext_vector_type(8)))  float    v8f;

#define CHUNK 4096   // j-chunk staged in LDS: 4096*(4+2+2) = 32 KB

// ---------------- Kernel A: theta = sigmoid(logits); e = exp(theta) split into f16 hi/lo
__global__ void cox_prep_kernel(const float* __restrict__ logits,
                                float* __restrict__ theta,
                                _Float16* __restrict__ ehi,
                                _Float16* __restrict__ elo,
                                int n) {
    int i = blockIdx.x * blockDim.x + threadIdx.x;
    if (i < n) {
        float t = 1.0f / (1.0f + expf(-logits[i]));
        float e = expf(t);
        _Float16 hi = (_Float16)e;
        float lo = e - (float)hi;
        theta[i] = t;
        ehi[i]   = hi;
        elo[i]   = (_Float16)lo;
    }
}

// ---------------- Kernel B: risk[i] = sum_j [Y[j] >= Y[i]] * e[j]  via WMMA
// One wave -> 16 output rows. Block = 8 waves = 128 rows. A = comparison mask (f16),
// B = e broadcast across N columns, C/D = f32 accumulators (all 16 columns identical).
// Two independent accumulator chains (e_hi / e_lo split) pipeline the WMMAs.
__global__ __launch_bounds__(256) void cox_risk_kernel(const float* __restrict__ Y,
                                                       const _Float16* __restrict__ ehi,
                                                       const _Float16* __restrict__ elo,
                                                       float* __restrict__ risk,
                                                       int n) {
    __shared__ float    sY[CHUNK];
    __shared__ _Float16 sHi[CHUNK];
    __shared__ _Float16 sLo[CHUNK];

    const int lane   = threadIdx.x & 31;
    const int wave   = threadIdx.x >> 5;
    const int ibase  = (blockIdx.x * 8 + wave) * 16;   // 16 rows per wave
    const int m      = lane & 15;                      // A-matrix row for this lane
    const int laneHi = lane >> 4;                      // K-half selector

    const int   yi_idx = ibase + m;
    const float Yi = (yi_idx < n) ? Y[yi_idx] : 3.0e38f;

    v8f acc_h = {};
    v8f acc_l = {};

    for (int c0 = 0; c0 < n; c0 += CHUNK) {
        // cooperative stage of this j-chunk into LDS (pad tail with e=0 => contributes 0)
        for (int t = threadIdx.x; t < CHUNK; t += blockDim.x) {
            int g = c0 + t;
            if (g < n) {
                sY[t]  = Y[g];
                sHi[t] = ehi[g];
                sLo[t] = elo[g];
            } else {
                sY[t]  = 0.0f;
                sHi[t] = (_Float16)0.0f;
                sLo[t] = (_Float16)0.0f;
            }
        }
        __syncthreads();

        // prefetch next chunk into the cache hierarchy while we compute this one
        int p0 = c0 + CHUNK;
        if (p0 < n) {
            int pi = p0 + (int)threadIdx.x * (CHUNK / 256);
            __builtin_prefetch(&Y[pi], 0, 0);
            __builtin_prefetch(&ehi[pi], 0, 0);
            __builtin_prefetch(&elo[pi], 0, 0);
        }

        for (int j = 0; j < CHUNK; j += 32) {
            v16h a, bh, bl;
#pragma unroll
            for (int t = 0; t < 16; ++t) {
                // 16-bit A 16x32 layout: VGPR v=t/2, half h=t&1
                // K = (v>=4 ? 16:0) + (laneHi ? 8:0) + (v&3)*2 + h
                const int v = t >> 1, h = t & 1;
                const int k = ((v >= 4) ? 16 : 0) + (laneHi ? 8 : 0) + ((v & 3) << 1) + h;
                a[t] = (sY[j + k] >= Yi) ? (_Float16)1.0f : (_Float16)0.0f;
                // 16-bit B 32x16 layout: lane-half selects K base; broadcast across N
                const int kb = k;
                bh[t] = sHi[j + kb];
                bl[t] = sLo[j + kb];
            }
            acc_h = __builtin_amdgcn_wmma_f32_16x16x32_f16(
                false, a, false, bh, (short)0, acc_h, false, false);
            acc_l = __builtin_amdgcn_wmma_f32_16x16x32_f16(
                false, a, false, bl, (short)0, acc_l, false, false);
        }
        __syncthreads();
    }

    // D layout: VGPR r -> row r (lanes 0-15) / row r+8 (lanes 16-31); columns identical.
    if (m == 0) {
        const int row0 = ibase + laneHi * 8;
        float r0 = acc_h[0] + acc_l[0], r1 = acc_h[1] + acc_l[1];
        float r2 = acc_h[2] + acc_l[2], r3 = acc_h[3] + acc_l[3];
        float r4 = acc_h[4] + acc_l[4], r5 = acc_h[5] + acc_l[5];
        float r6 = acc_h[6] + acc_l[6], r7 = acc_h[7] + acc_l[7];
        if (row0 + 8 <= n) {
            float4 lo4 = make_float4(r0, r1, r2, r3);
            float4 hi4 = make_float4(r4, r5, r6, r7);
            *(float4*)(risk + row0)     = lo4;   // global_store_b128
            *(float4*)(risk + row0 + 4) = hi4;   // global_store_b128
        } else {
            float rr[8] = {r0, r1, r2, r3, r4, r5, r6, r7};
            for (int r = 0; r < 8 && row0 + r < n; ++r) risk[row0 + r] = rr[r];
        }
    }
}

// ---------------- Kernel C: loss = -(1/n) * sum_i c[i] * (theta[i] - log(risk[i]))
__global__ __launch_bounds__(256) void cox_finish_kernel(const float* __restrict__ theta,
                                                         const int* __restrict__ c,
                                                         const float* __restrict__ risk,
                                                         float* __restrict__ out,
                                                         int n) {
    __shared__ float red[256];
    float s = 0.0f;
    for (int i = threadIdx.x; i < n; i += blockDim.x) {
        if (c[i]) s += theta[i] - logf(risk[i]);
    }
    red[threadIdx.x] = s;
    __syncthreads();
    for (int off = 128; off > 0; off >>= 1) {
        if ((int)threadIdx.x < off) red[threadIdx.x] += red[threadIdx.x + off];
        __syncthreads();
    }
    if (threadIdx.x == 0) out[0] = -red[0] / (float)n;
}

extern "C" void kernel_launch(void* const* d_in, const int* in_sizes, int n_in,
                              void* d_out, int out_size, void* d_ws, size_t ws_size,
                              hipStream_t stream) {
    const float* Y      = (const float*)d_in[0];
    const int*   c      = (const int*)d_in[1];
    const float* logits = (const float*)d_in[2];
    const int n = in_sizes[0];

    // workspace layout: theta f32[n] | ehi f16[n] | elo f16[n] | risk f32[n]  (n*12 bytes)
    char* base = (char*)d_ws;
    float*    theta = (float*)base;
    _Float16* ehi   = (_Float16*)(base + (size_t)n * 4);
    _Float16* elo   = ehi + n;
    float*    risk  = (float*)(base + (size_t)n * 8);

    cox_prep_kernel<<<(n + 255) / 256, 256, 0, stream>>>(logits, theta, ehi, elo, n);

    const int rowTiles = (n + 15) / 16;            // 16 rows per wave
    const int blocks   = (rowTiles + 7) / 8;       // 8 waves per block
    cox_risk_kernel<<<blocks, 256, 0, stream>>>(Y, ehi, elo, risk, n);

    cox_finish_kernel<<<1, 256, 0, stream>>>(theta, c, risk, (float*)d_out, n);
}
